// STSSTransformation_28329604284784
// MI455X (gfx1250) — matrix operand order repaired and checked by reading, not compiled, not run on hardware
//
#include <hip/hip_runtime.h>
#include <hip/hip_bf16.h>

// Problem constants (from reference)
#define B_   4
#define T_   8
#define C_   512
#define H_   28
#define W_   28
#define HW_  (H_*W_)
#define L_   5
#define P_   9
#define NPIX (B_*T_*HW_)
#define EPSF 1e-6f

// Phase-2 tiling
#define KC        32    // K (channel) chunk per WMMA
#define NCHUNK    (C_/KC)
#define QS_STRIDE 40    // halves per query-w row in LDS
#define KS_STRIDE 40    // halves per key-off row in LDS
#define KS_W      48    // key buffer width in off = w' + 4
#define TRASH_OFF (W_*QS_STRIDE)              // 1120 halves (16B aligned)
#define KS_BASE   (W_*QS_STRIDE + 8)          // 1128 halves (16B aligned)
#define BUF_HALVES (KS_BASE + P_*KS_W*KS_STRIDE)  // 18408 halves per buffer

typedef __attribute__((ext_vector_type(16))) _Float16 v16h;
typedef __attribute__((ext_vector_type(8)))  _Float16 v8h;
typedef __attribute__((ext_vector_type(8)))  float    v8f;
typedef __attribute__((ext_vector_type(4)))  int      i4;

// ---- CDNA5 async global->LDS path (ASYNCcnt-tracked) ----
#if defined(__has_builtin)
#if __has_builtin(__builtin_amdgcn_global_load_async_to_lds_b128)
#define ASYNC_LDS 1
#else
#define ASYNC_LDS 0
#endif
#else
#define ASYNC_LDS 0
#endif

typedef __attribute__((address_space(1))) i4* glb_i4p;
typedef __attribute__((address_space(3))) i4* lds_i4p;

__device__ __forceinline__ void wait_async0() {
#if defined(__has_builtin) && __has_builtin(__builtin_amdgcn_s_wait_asynccnt)
  __builtin_amdgcn_s_wait_asynccnt(0);
#else
  asm volatile("s_wait_asynccnt 0" ::: "memory");
#endif
}
__device__ __forceinline__ void wait_async5() {
#if defined(__has_builtin) && __has_builtin(__builtin_amdgcn_s_wait_asynccnt)
  __builtin_amdgcn_s_wait_asynccnt(5);
#else
  asm volatile("s_wait_asynccnt 5" ::: "memory");
#endif
}

// ---------------------------------------------------------------------------
// Phase 1a: per-pixel channel L2 norm reciprocal
// ---------------------------------------------------------------------------
__global__ void rnorm_kernel(const float* __restrict__ x, float* __restrict__ rnorm) {
  int pix = blockIdx.x * 256 + threadIdx.x;
  if (pix >= NPIX) return;
  int bt = pix / HW_;
  int hw = pix - bt * HW_;
  const float* p = x + (size_t)bt * C_ * HW_ + hw;
  float s = 0.f;
#pragma unroll 4
  for (int c = 0; c < C_; ++c) { float v = p[(size_t)c * HW_]; s += v * v; }
  rnorm[pix] = rsqrtf(s + EPSF);
}

// zero-fill the 512-half dummy source row (read for OOB key rows)
__global__ void zfill_kernel(unsigned* __restrict__ zr32) {
  zr32[threadIdx.x] = 0u;   // 256 threads x 4B = 1KB = 512 halves
}

// ---------------------------------------------------------------------------
// Phase 1b: xn[bt][h][w][c] = f16( x[bt][c][h][w] * rnorm[bt][h][w] )
// ---------------------------------------------------------------------------
__global__ void xn_kernel(const float* __restrict__ x, const float* __restrict__ rnorm,
                          _Float16* __restrict__ xn) {
  __shared__ _Float16 sm[C_ * W_];
  __shared__ float    rn[W_];
  const int h  = blockIdx.x;
  const int bt = blockIdx.y;
  const int tid = threadIdx.x;
  if (tid < W_) rn[tid] = rnorm[bt * HW_ + h * W_ + tid];
  __syncthreads();
  const float* p = x + (size_t)bt * C_ * HW_ + h * W_;
  for (int i = tid; i < C_ * W_; i += 256) {
    int c = i / W_, w = i - c * W_;
    sm[i] = (_Float16)(p[(size_t)c * HW_ + w] * rn[w]);
  }
  __syncthreads();
  _Float16* q = xn + ((size_t)bt * H_ + h) * W_ * C_;
  for (int i = tid; i < C_ * W_; i += 256) {
    int w = i / C_, c = i - w * C_;
    q[i] = sm[c * W_ + w];
  }
}

// ---------------------------------------------------------------------------
// LDS fragment loaders matching CDNA5 16-bit WMMA VGPR layouts
// ---------------------------------------------------------------------------
__device__ __forceinline__ v16h ld_contig16(const _Float16* p) {
  v8h lo = *(const v8h*)p;
  v8h hi = *(const v8h*)(p + 8);
  v16h r;
#pragma unroll
  for (int i = 0; i < 8; ++i) { r[i] = lo[i]; r[i + 8] = hi[i]; }
  return r;
}
__device__ __forceinline__ v16h ld_afrag(const _Float16* row, int sh) {
  v8h lo = *(const v8h*)(row + sh);
  v8h hi = *(const v8h*)(row + 16 + sh);
  v16h r;
#pragma unroll
  for (int i = 0; i < 8; ++i) { r[i] = lo[i]; r[i + 8] = hi[i]; }
  return r;
}

// ---------------------------------------------------------------------------
// Phase 2: banded Gram via v_wmma_f32_16x16x32_f16, double-buffered async pipeline.
// One block = (b,t,l,h). 9 waves, wave u handles key row h' = h+u-4.
// ---------------------------------------------------------------------------
__global__ __launch_bounds__(288, 1)
void corr_kernel(const _Float16* __restrict__ xn, const _Float16* __restrict__ zrow,
                 float* __restrict__ out) {
  __shared__ __align__(16) _Float16 BUF[2][BUF_HALVES];   // 2 x 36816 B

  const int h  = blockIdx.x;
  const int l  = blockIdx.y;
  const int bt = blockIdx.z;
  const int t  = bt % T_;
  const int tp = t + l - 2;                 // key time index
  const bool tvalid = (tp >= 0) && (tp < T_);
  const int btp = bt + (l - 2);             // same b when tvalid
  const int tid = threadIdx.x;

  // Zero both buffers once (halo columns stay zero; data rows re-filled per chunk)
  for (int i = tid; i < 2 * BUF_HALVES; i += 288) BUF[0][i] = (_Float16)0.f;
  __syncthreads();

  const int u    = tid >> 5;
  const int lane = tid & 31;
  const int nl   = lane & 15;
  const int sh   = (lane < 16) ? 0 : 8;
  const int kh   = (lane < 16) ? 0 : 16;

  // ---- precompute branch-free fill geometry (5 async loads per thread) ----
  _Float16* xnm  = const_cast<_Float16*>(xn);
  _Float16* zr   = const_cast<_Float16*>(zrow);
  _Float16* qrow = xnm + ((size_t)bt * H_ + h) * W_ * C_;

  const int qt = (tid < 112) ? tid : 0;
  const int qw = qt >> 2, qk = (qt & 3) * 8;
  _Float16* srcQ = qrow + (size_t)qw * C_ + qk;
  const int dstQ = (tid < 112) ? (qw * QS_STRIDE + qk) : TRASH_OFF;

  _Float16* srcK[4]; int dstK[4];
#pragma unroll
  for (int j = 0; j < 4; ++j) {
    int i  = j * 288 + tid;
    bool act = (i < P_ * 112);
    int ii = act ? i : 0;
    int uu = ii / 112, r = ii - uu * 112;
    int wp = r >> 2, kq = (r & 3) * 8;
    int hp = h + uu - 4;
    bool valid = act && tvalid && (hp >= 0) && (hp < H_);
    int hpc = hp < 0 ? 0 : (hp > H_ - 1 ? H_ - 1 : hp);
    srcK[j] = valid ? (xnm + ((size_t)(tvalid ? btp : 0) * H_ + hpc) * W_ * C_
                            + (size_t)wp * C_ + kq)
                    : (zr + kq);
    dstK[j] = act ? (KS_BASE + (uu * KS_W + wp + 4) * KS_STRIDE + kq) : TRASH_OFF;
  }

  v8f acc00 = {}, acc01 = {}, acc10 = {}, acc11 = {};

#if ASYNC_LDS
  // ---- issue chunk: exactly 5 wave-uniform async instructions ----
  #define ISSUE_CHUNK(bufi, kc)                                                  \
    do {                                                                         \
      __builtin_amdgcn_global_load_async_to_lds_b128(                            \
          (glb_i4p)(srcQ + (kc)), (lds_i4p)&BUF[bufi][dstQ], 0, 0);              \
      _Pragma("unroll")                                                          \
      for (int j = 0; j < 4; ++j)                                                \
        __builtin_amdgcn_global_load_async_to_lds_b128(                          \
            (glb_i4p)(srcK[j] + (kc)), (lds_i4p)&BUF[bufi][dstK[j]], 0, 0);      \
    } while (0)
#else
  // fallback: register-staged synchronous copy (batched loads, one wait)
  #define ISSUE_CHUNK(bufi, kc)                                                  \
    do {                                                                         \
      v8h qv = *(const v8h*)(srcQ + (kc));                                       \
      v8h kv[4];                                                                 \
      _Pragma("unroll")                                                          \
      for (int j = 0; j < 4; ++j) kv[j] = *(const v8h*)(srcK[j] + (kc));         \
      *(v8h*)&BUF[bufi][dstQ] = qv;                                              \
      _Pragma("unroll")                                                          \
      for (int j = 0; j < 4; ++j) *(v8h*)&BUF[bufi][dstK[j]] = kv[j];            \
    } while (0)
  #define wait_async5() ((void)0)
  #undef  ASYNC_LDS
  #define ASYNC_LDS 0
#endif

  // ---- compute one chunk from buffer `buf` ----
  #define COMPUTE_CHUNK(buf)                                                     \
    do {                                                                         \
      const _Float16* Qb = &BUF[buf][0];                                         \
      const _Float16* Kb = &BUF[buf][KS_BASE] + (size_t)u * KS_W * KS_STRIDE;    \
      v16h a0 = ld_contig16(&Qb[(0  + nl) * QS_STRIDE + kh]);                    \
      v16h a1 = ld_contig16(&Qb[(12 + nl) * QS_STRIDE + kh]);                    \
      v16h b00 = ld_contig16(&Kb[(0  + nl) * KS_STRIDE + kh]);                   \
      v16h b16 = ld_contig16(&Kb[(16 + nl) * KS_STRIDE + kh]);                   \
      v16h b12 = ld_contig16(&Kb[(12 + nl) * KS_STRIDE + kh]);                   \
      v16h b28 = ld_contig16(&Kb[(28 + nl) * KS_STRIDE + kh]);                   \
      acc00 = __builtin_amdgcn_wmma_f32_16x16x32_f16(false, a0, false, b00, (short)0, acc00, false, false); \
      acc01 = __builtin_amdgcn_wmma_f32_16x16x32_f16(false, a0, false, b16, (short)0, acc01, false, false); \
      acc10 = __builtin_amdgcn_wmma_f32_16x16x32_f16(false, a1, false, b12, (short)0, acc10, false, false); \
      acc11 = __builtin_amdgcn_wmma_f32_16x16x32_f16(false, a1, false, b28, (short)0, acc11, false, false); \
    } while (0)

  // ---- software pipeline: copy chunk n+1 overlaps WMMA of chunk n ----
  ISSUE_CHUNK(0, 0);
#pragma unroll 1
  for (int n = 0; n < NCHUNK - 1; ++n) {
    ISSUE_CHUNK((n + 1) & 1, (n + 1) * KC);
#if ASYNC_LDS
    wait_async5();          // in-order completion => chunk n fully in LDS
#endif
    __syncthreads();        // all waves see chunk n data
    COMPUTE_CHUNK(n & 1);
    __syncthreads();        // buffer n&1 free for chunk n+2
  }
  wait_async0();
  __syncthreads();
  COMPUTE_CHUNK((NCHUNK - 1) & 1);

  // ---- writeout: D[M,N] -> out[..., w = m0+M, l, u, v = 16*nt + N - M] ----
  const int rbase = (lane < 16) ? 0 : 8;
  const size_t obase = ((size_t)bt * H_ + h) * W_ * (size_t)(L_ * P_ * P_)
                     + (size_t)l * (P_ * P_) + (size_t)u * P_;
#pragma unroll
  for (int r = 0; r < 8; ++r) {
    int M  = rbase + r;
    int v0 = nl - M;
    int v1 = 16 + nl - M;
    if (v0 >= 0 && v0 < P_) {
      out[obase + (size_t)(0  + M) * (L_ * P_ * P_) + v0] = acc00[r];
      out[obase + (size_t)(12 + M) * (L_ * P_ * P_) + v0] = acc10[r];
    }
    if (v1 >= 0 && v1 < P_) {
      out[obase + (size_t)(0  + M) * (L_ * P_ * P_) + v1] = acc01[r];
      out[obase + (size_t)(12 + M) * (L_ * P_ * P_) + v1] = acc11[r];
    }
  }
}

// ---------------------------------------------------------------------------
extern "C" void kernel_launch(void* const* d_in, const int* in_sizes, int n_in,
                              void* d_out, int out_size, void* d_ws, size_t ws_size,
                              hipStream_t stream) {
  (void)in_sizes; (void)n_in; (void)out_size; (void)ws_size;
  const float* x = (const float*)d_in[0];
  float* out = (float*)d_out;

  // workspace: xn f16 (25.69 MB) | rnorm f32 (100 KB) | zero row (1 KB)
  _Float16* xn    = (_Float16*)d_ws;
  size_t xn_bytes = (size_t)NPIX * C_ * sizeof(_Float16);
  float*    rnorm = (float*)((char*)d_ws + xn_bytes);
  _Float16* zrow  = (_Float16*)((char*)d_ws + xn_bytes + (size_t)NPIX * sizeof(float));

  rnorm_kernel<<<(NPIX + 255) / 256, 256, 0, stream>>>(x, rnorm);
  zfill_kernel<<<1, 256, 0, stream>>>((unsigned*)zrow);
  xn_kernel<<<dim3(H_, B_ * T_), 256, 0, stream>>>(x, rnorm, xn);
  corr_kernel<<<dim3(H_, L_, B_ * T_), 288, 0, stream>>>(xn, zrow, out);
}
